// SwinTransformerBlock_2525440770316
// MI455X (gfx1250) — matrix operand order repaired and checked by reading, not compiled, not run on hardware
//
#include <hip/hip_runtime.h>
#include <hip/hip_bf16.h>
#include <math.h>

// ---------------- Problem constants ----------------
#define Bsz   32
#define Hdim  56
#define Wdim  56
#define Cdim  384
#define NHead 12
#define HD    32          // head dim = C/NH
#define WS7   7
#define SS3   3
#define Nwin  49          // WS*WS
#define NWb   64          // windows per batch
#define BNW   2048        // B*NW
#define Mrows 100352      // B*H*W
#define C3    1152        // 3*C
#define CM    1536        // MLP*C

typedef _Float16 v16h __attribute__((ext_vector_type(16)));
typedef _Float16 v8h  __attribute__((ext_vector_type(8)));
typedef float    v8f  __attribute__((ext_vector_type(8)));

__device__ __forceinline__ v16h ld_frag(const _Float16* p0, const _Float16* p1) {
  union { v16h v; v8h h[2]; } u;
  u.h[0] = *(const v8h*)p0;
  u.h[1] = *(const v8h*)p1;
  return u.v;
}

__device__ __forceinline__ v8f wmma_f16(v16h a, v16h b, v8f c) {
  return __builtin_amdgcn_wmma_f32_16x16x32_f16(false, a, false, b, (short)0, c, false, false);
}

// Async Global -> LDS copy (gfx1250 VGLOBAL async op, tracked by ASYNCcnt).
// vdst = LDS byte offset (low 32 bits of a generic shared pointer), vaddr = 64b address.
__device__ __forceinline__ void async_ld_b128(unsigned int lds_off, const void* g) {
  asm volatile("global_load_async_to_lds_b128 %0, %1, off"
               :: "v"(lds_off), "v"((unsigned long long)(size_t)g)
               : "memory");
}
__device__ __forceinline__ void wait_async0() {
  asm volatile("s_wait_asynccnt 0x0" ::: "memory");
}
__device__ __forceinline__ unsigned int lds_off_of(const void* p) {
  return (unsigned int)(size_t)p;
}

// ---------------- f32 -> f16 weight convert ----------------
__global__ void __launch_bounds__(256) cvt_kernel(const float* __restrict__ src,
                                                  _Float16* __restrict__ dst, int n) {
  int i = blockIdx.x * 256 + threadIdx.x;
  if (i < n) dst[i] = (_Float16)src[i];
}

// ---------------- LayerNorm (optionally shift+window-partition the output) ----------------
__global__ void __launch_bounds__(128) ln_kernel(const float* __restrict__ X,
                                                 const float* __restrict__ g,
                                                 const float* __restrict__ be,
                                                 _Float16* __restrict__ out,
                                                 int window_map) {
  __shared__ float r1[128], r2[128];
  const int row = blockIdx.x, tid = threadIdx.x;
  const float* xr = X + (size_t)row * Cdim;
  float v0 = xr[tid], v1 = xr[tid + 128], v2 = xr[tid + 256];
  r1[tid] = v0 + v1 + v2;
  r2[tid] = v0 * v0 + v1 * v1 + v2 * v2;
  __syncthreads();
  for (int o = 64; o > 0; o >>= 1) {
    if (tid < o) { r1[tid] += r1[tid + o]; r2[tid] += r2[tid + o]; }
    __syncthreads();
  }
  float mean = r1[0] * (1.f / Cdim);
  float var  = r2[0] * (1.f / Cdim) - mean * mean;
  float rstd = rsqrtf(var + 1e-5f);
  size_t base;
  if (window_map) {
    int b   = row / (Hdim * Wdim);
    int pos = row - b * (Hdim * Wdim);
    int hh = pos / Wdim, ww = pos - (pos / Wdim) * Wdim;
    int hp = hh + (Hdim - SS3); if (hp >= Hdim) hp -= Hdim;   // shifted coord
    int wp = ww + (Wdim - SS3); if (wp >= Wdim) wp -= Wdim;
    int widx = b * NWb + (hp / WS7) * 8 + (wp / WS7);
    int n    = (hp % WS7) * WS7 + (wp % WS7);
    base = ((size_t)widx * Nwin + n) * Cdim;
  } else {
    base = (size_t)row * Cdim;
  }
  out[base + tid]       = (_Float16)((v0 - mean) * rstd * g[tid]       + be[tid]);
  out[base + tid + 128] = (_Float16)((v1 - mean) * rstd * g[tid + 128] + be[tid + 128]);
  out[base + tid + 256] = (_Float16)((v2 - mean) * rstd * g[tid + 256] + be[tid + 256]);
}

// ---------------- WMMA GEMM: C[M,Nn] = A[M,K] @ W[Nn,K]^T (+epilogue) ----------------
// Block = 256 threads (8 waves, wave32). Block tile 128x128, wave tile 32x64.
// K chunks of 32 staged via async Global->LDS into ping-pong buffers:
// next chunk's async copies overlap current chunk's WMMAs; one barrier per chunk.
// MODE 0: f16 out + bias (QKV)
// MODE 1: f32 out = resid[dst] + acc + bias, dst = window-reverse + unshift (proj)
// MODE 2: f16 out = gelu(acc + bias) (FC1)
// MODE 3: f32 out = resid + acc + bias, same row order (FC2 -> d_out)
template <int MODE>
__global__ void __launch_bounds__(256) gemm_kernel(const _Float16* __restrict__ A,
                                                   const _Float16* __restrict__ W,
                                                   const float* __restrict__ bias,
                                                   void* __restrict__ out,
                                                   const float* __restrict__ resid,
                                                   int M, int Nn, int K) {
  __shared__ _Float16 As[2][128 * 40];
  __shared__ _Float16 Bs[2][128 * 40];
  const int tid    = threadIdx.x;
  const int wave   = tid >> 5;
  const int lane   = tid & 31;
  const int lm     = lane & 15;
  const int ah     = lane >> 4;     // which half of the wave
  const int wm     = wave >> 1;     // 0..3
  const int wn     = wave & 1;      // 0..1
  const int blockN = blockIdx.x * 128;
  const int blockM = blockIdx.y * 128;
  const int koffA  = ah * 8;
  const int koffB  = ah * 16;

  // per-thread staging assignment: 2 A-segments + 2 B-segments (16B each) per chunk
  const int r0 = tid >> 2,          s0 = tid & 3;        // idx = tid
  const int r1 = (tid + 256) >> 2,  s1 = tid & 3;        // idx = tid + 256

  auto stage = [&](int buf, int kc) {
    async_ld_b128(lds_off_of(&As[buf][r0 * 40 + s0 * 8]),
                  &A[(size_t)(blockM + r0) * K + kc + s0 * 8]);
    async_ld_b128(lds_off_of(&Bs[buf][r0 * 40 + s0 * 8]),
                  &W[(size_t)(blockN + r0) * K + kc + s0 * 8]);
    async_ld_b128(lds_off_of(&As[buf][r1 * 40 + s1 * 8]),
                  &A[(size_t)(blockM + r1) * K + kc + s1 * 8]);
    async_ld_b128(lds_off_of(&Bs[buf][r1 * 40 + s1 * 8]),
                  &W[(size_t)(blockN + r1) * K + kc + s1 * 8]);
  };

  v8f acc[2][4] = {};

  stage(0, 0);
  wait_async0();
  __syncthreads();

  int cur = 0;
  for (int kc = 0; kc < K; kc += 32) {
    if (kc + 32 < K) stage(cur ^ 1, kc + 32);   // prefetch next chunk (overlaps WMMAs)

    v16h af[2], bf[4];
#pragma unroll
    for (int ti = 0; ti < 2; ++ti) {
      const _Float16* p = &As[cur][(wm * 32 + ti * 16 + lm) * 40];
      af[ti] = ld_frag(p + koffA, p + 16 + koffA);
    }
#pragma unroll
    for (int tj = 0; tj < 4; ++tj) {
      const _Float16* p = &Bs[cur][(wn * 64 + tj * 16 + lm) * 40];
      bf[tj] = ld_frag(p + koffA, p + 16 + koffA);
    }
#pragma unroll
    for (int ti = 0; ti < 2; ++ti)
#pragma unroll
      for (int tj = 0; tj < 4; ++tj)
        acc[ti][tj] = wmma_f16(af[ti], bf[tj], acc[ti][tj]);

    wait_async0();      // own async copies for next chunk have landed in LDS
    __syncthreads();    // everyone's have
    cur ^= 1;
  }

#pragma unroll
  for (int ti = 0; ti < 2; ++ti) {
#pragma unroll
    for (int tj = 0; tj < 4; ++tj) {
      int row0 = blockM + wm * 32 + ti * 16 + ah * 8;
      int col  = blockN + wn * 64 + tj * 16 + lm;
      float bc = bias[col];
#pragma unroll
      for (int r = 0; r < 8; ++r) {
        int row = row0 + r;
        float v = acc[ti][tj][r] + bc;
        if constexpr (MODE == 0) {
          ((_Float16*)out)[(size_t)row * Nn + col] = (_Float16)v;
        } else if constexpr (MODE == 2) {
          float g = 0.5f * v * (1.f + erff(v * 0.70710678118f));
          ((_Float16*)out)[(size_t)row * Nn + col] = (_Float16)g;
        } else if constexpr (MODE == 1) {
          int wdx = row / Nwin, n = row - wdx * Nwin;
          int bb = wdx >> 6, nw = wdx & 63;
          int hp = (nw >> 3) * WS7 + n / WS7;
          int wp = (nw & 7) * WS7 + n % WS7;
          int hh = hp + SS3; if (hh >= Hdim) hh -= Hdim;  // reverse shift
          int ww = wp + SS3; if (ww >= Wdim) ww -= Wdim;
          size_t dst = ((size_t)bb * (Hdim * Wdim) + hh * Wdim + ww) * Cdim + col;
          ((float*)out)[dst] = resid[dst] + v;
        } else {  // MODE == 3
          size_t dst = (size_t)row * Cdim + col;
          ((float*)out)[dst] = resid[dst] + v;
        }
      }
    }
  }
}

// ---------------- Attention: one wave per (window, head) ----------------
__global__ void __launch_bounds__(32) attn_kernel(const _Float16* __restrict__ qkv,
                                                  const float* __restrict__ rpb,
                                                  const int* __restrict__ rpi,
                                                  const float* __restrict__ mask,
                                                  _Float16* __restrict__ o) {
  __shared__ _Float16 qs[64 * 40];
  __shared__ _Float16 ks[64 * 40];
  __shared__ _Float16 vs[32 * 72];   // v transposed: [d][m], K-major for B-frag
  __shared__ _Float16 ps[64 * 72];   // softmax probs, row-major [m][n]
  const int bid  = blockIdx.x;
  const int w    = bid / NHead;
  const int h    = bid - w * NHead;
  const int lane = threadIdx.x;
  const int lm   = lane & 15;
  const int ah   = lane >> 4;

  // stage q, k (zero-pad rows 49..63) and v transposed
  for (int m = lane; m < 64; m += 32) {
    float4* qd = (float4*)&qs[m * 40];
    float4* kd = (float4*)&ks[m * 40];
    if (m < Nwin) {
      const float4* qsrc = (const float4*)(qkv + (size_t)(w * Nwin + m) * C3 + h * HD);
      const float4* ksrc = (const float4*)(qkv + (size_t)(w * Nwin + m) * C3 + Cdim + h * HD);
#pragma unroll
      for (int s = 0; s < 4; ++s) { qd[s] = qsrc[s]; kd[s] = ksrc[s]; }
      const _Float16* vsrc = qkv + (size_t)(w * Nwin + m) * C3 + 2 * Cdim + h * HD;
#pragma unroll
      for (int d = 0; d < 32; ++d) vs[d * 72 + m] = vsrc[d];
    } else {
      float4 z = {0.f, 0.f, 0.f, 0.f};
#pragma unroll
      for (int s = 0; s < 4; ++s) { qd[s] = z; kd[s] = z; }
#pragma unroll
      for (int d = 0; d < 32; ++d) vs[d * 72 + m] = (_Float16)0.f;
    }
  }
  __syncthreads();

  // s = q @ k^T  (64x64 padded, K = 32 = one WMMA per tile)
  v8f sacc[4][4] = {};
  {
    v16h af[4], bf[4];
#pragma unroll
    for (int ti = 0; ti < 4; ++ti) {
      const _Float16* p = &qs[(ti * 16 + lm) * 40];
      af[ti] = ld_frag(p + ah * 8, p + 16 + ah * 8);
    }
#pragma unroll
    for (int tj = 0; tj < 4; ++tj) {
      const _Float16* p = &ks[(tj * 16 + lm) * 40];
      bf[tj] = ld_frag(p + ah * 8, p + 16 + ah * 8);
    }
#pragma unroll
    for (int ti = 0; ti < 4; ++ti)
#pragma unroll
      for (int tj = 0; tj < 4; ++tj)
        sacc[ti][tj] = wmma_f16(af[ti], bf[tj], sacc[ti][tj]);
  }

  // scale + rel-pos bias + mask + row softmax (rows live across 16 lanes)
  const float  scale = 0.17677669529663687f;  // 1/sqrt(32)
  const float* mw    = mask + (size_t)(w & (NWb - 1)) * (Nwin * Nwin);
#pragma unroll
  for (int ti = 0; ti < 4; ++ti) {
#pragma unroll
    for (int r = 0; r < 8; ++r) {
      int m = ti * 16 + ah * 8 + r;
      float sv[4];
#pragma unroll
      for (int tj = 0; tj < 4; ++tj) {
        int n = tj * 16 + lm;
        float val = sacc[ti][tj][r] * scale;
        if (n >= Nwin)      val = -1e30f;                       // padded cols vanish
        else if (m < Nwin)  val += rpb[rpi[m * Nwin + n] * NHead + h] + mw[m * Nwin + n];
        else                val = 0.f;                          // padded rows: uniform
        sv[tj] = val;
      }
      float mx = fmaxf(fmaxf(sv[0], sv[1]), fmaxf(sv[2], sv[3]));
      for (int d = 1; d <= 8; d <<= 1) mx = fmaxf(mx, __shfl_xor(mx, d, 32));
      float sum = 0.f;
#pragma unroll
      for (int tj = 0; tj < 4; ++tj) { sv[tj] = expf(sv[tj] - mx); sum += sv[tj]; }
      for (int d = 1; d <= 8; d <<= 1) sum += __shfl_xor(sum, d, 32);
      float inv = 1.f / sum;
#pragma unroll
      for (int tj = 0; tj < 4; ++tj)
        ps[m * 72 + tj * 16 + lm] = (_Float16)(sv[tj] * inv);
    }
  }
  __syncthreads();

  // o = p @ v  (64x32, K = 64 -> two 32-chunks)
  v8f oacc[4][2] = {};
#pragma unroll
  for (int kc = 0; kc < 64; kc += 32) {
    v16h pa[4], vb[2];
#pragma unroll
    for (int ti = 0; ti < 4; ++ti) {
      const _Float16* p = &ps[(ti * 16 + lm) * 72 + kc];
      pa[ti] = ld_frag(p + ah * 8, p + 16 + ah * 8);
    }
#pragma unroll
    for (int dj = 0; dj < 2; ++dj) {
      const _Float16* p = &vs[(dj * 16 + lm) * 72 + kc];
      vb[dj] = ld_frag(p + ah * 8, p + 16 + ah * 8);
    }
#pragma unroll
    for (int ti = 0; ti < 4; ++ti)
#pragma unroll
      for (int dj = 0; dj < 2; ++dj)
        oacc[ti][dj] = wmma_f16(pa[ti], vb[dj], oacc[ti][dj]);
  }
#pragma unroll
  for (int ti = 0; ti < 4; ++ti)
#pragma unroll
    for (int dj = 0; dj < 2; ++dj)
#pragma unroll
      for (int r = 0; r < 8; ++r) {
        int m = ti * 16 + ah * 8 + r;
        if (m < Nwin)
          o[(size_t)(w * Nwin + m) * Cdim + h * HD + dj * 16 + lm] = (_Float16)oacc[ti][dj][r];
      }
}

// ---------------- Host orchestration ----------------
extern "C" void kernel_launch(void* const* d_in, const int* in_sizes, int n_in,
                              void* d_out, int out_size, void* d_ws, size_t ws_size,
                              hipStream_t stream) {
  const float* x      = (const float*)d_in[0];
  const float* ln1_w  = (const float*)d_in[1];
  const float* ln1_b  = (const float*)d_in[2];
  const float* qkv_w  = (const float*)d_in[3];
  const float* qkv_b  = (const float*)d_in[4];
  const float* rpb    = (const float*)d_in[5];
  const float* proj_w = (const float*)d_in[6];
  const float* proj_b = (const float*)d_in[7];
  const float* ln2_w  = (const float*)d_in[8];
  const float* ln2_b  = (const float*)d_in[9];
  const float* fc1_w  = (const float*)d_in[10];
  const float* fc1_b  = (const float*)d_in[11];
  const float* fc2_w  = (const float*)d_in[12];
  const float* fc2_b  = (const float*)d_in[13];
  const float* mask   = (const float*)d_in[14];
  const int*   rpi    = (const int*)d_in[15];

  char*  ws  = (char*)d_ws;
  size_t off = 0;
  auto alloc = [&](size_t bytes) -> void* {
    void* p = ws + off;
    off += (bytes + 255) & ~(size_t)255;
    return p;
  };
  _Float16* qkvw_h = (_Float16*)alloc((size_t)C3  * Cdim * 2);
  _Float16* projw_h= (_Float16*)alloc((size_t)Cdim* Cdim * 2);
  _Float16* fc1w_h = (_Float16*)alloc((size_t)CM  * Cdim * 2);
  _Float16* fc2w_h = (_Float16*)alloc((size_t)Cdim* CM   * 2);
  _Float16* win_h  = (_Float16*)alloc((size_t)Mrows * Cdim * 2); // LN1 windows, reused for LN2
  _Float16* big_h  = (_Float16*)alloc((size_t)Mrows * CM   * 2); // qkv, reused for fc1 out
  _Float16* o_h    = (_Float16*)alloc((size_t)Mrows * Cdim * 2);
  float*    x1_f   = (float*)   alloc((size_t)Mrows * Cdim * 4);

  // 1) weights to f16
  cvt_kernel<<<(C3 * Cdim + 255) / 256, 256, 0, stream>>>(qkv_w,  qkvw_h,  C3 * Cdim);
  cvt_kernel<<<(Cdim * Cdim + 255) / 256, 256, 0, stream>>>(proj_w, projw_h, Cdim * Cdim);
  cvt_kernel<<<(CM * Cdim + 255) / 256, 256, 0, stream>>>(fc1_w,  fc1w_h,  CM * Cdim);
  cvt_kernel<<<(Cdim * CM + 255) / 256, 256, 0, stream>>>(fc2_w,  fc2w_h,  Cdim * CM);

  // 2) LN1 + cyclic shift + window partition -> win_h
  ln_kernel<<<Mrows, 128, 0, stream>>>(x, ln1_w, ln1_b, win_h, 1);

  // 3) QKV GEMM -> big_h
  gemm_kernel<0><<<dim3(C3 / 128, Mrows / 128), 256, 0, stream>>>(
      win_h, qkvw_h, qkv_b, big_h, nullptr, Mrows, C3, Cdim);

  // 4) windowed attention -> o_h
  attn_kernel<<<BNW * NHead, 32, 0, stream>>>(big_h, rpb, rpi, mask, o_h);

  // 5) proj GEMM + window-reverse/unshift + residual(x) -> x1_f
  gemm_kernel<1><<<dim3(Cdim / 128, Mrows / 128), 256, 0, stream>>>(
      o_h, projw_h, proj_b, x1_f, x, Mrows, Cdim, Cdim);

  // 6) LN2 -> win_h (reuse)
  ln_kernel<<<Mrows, 128, 0, stream>>>(x1_f, ln2_w, ln2_b, win_h, 0);

  // 7) FC1 GEMM + GELU -> big_h (reuse)
  gemm_kernel<2><<<dim3(CM / 128, Mrows / 128), 256, 0, stream>>>(
      win_h, fc1w_h, fc1_b, big_h, nullptr, Mrows, CM, Cdim);

  // 8) FC2 GEMM + residual(x1) -> d_out
  gemm_kernel<3><<<dim3(Cdim / 128, Mrows / 128), 256, 0, stream>>>(
      big_h, fc2w_h, fc2_b, d_out, x1_f, Mrows, Cdim, CM);
}
